// LinearTimeSelfAttention1D_76879914599082
// MI455X (gfx1250) — matrix operand order repaired and checked by reading, not compiled, not run on hardware
//
#include <hip/hip_runtime.h>
#include <hip/hip_bf16.h>
#include <math.h>

// ---------------------------------------------------------------------------
// LinearTimeSelfAttention1D on MI455X (gfx1250, wave32, WMMA bf16 16x16x32)
// ---------------------------------------------------------------------------

typedef __bf16 bf16;
typedef __attribute__((ext_vector_type(16))) __bf16 v16bf;
typedef __attribute__((ext_vector_type(8)))  __bf16 v8bf;
typedef __attribute__((ext_vector_type(4)))  __bf16 v4bf;
typedef __attribute__((ext_vector_type(8)))  float  v8f;

#define DIM     256
#define L       32768
#define HEADS   8
#define DHEAD   64
#define HID     512
#define GROUPS  32
#define CPG     8          // DIM / GROUPS
#define SPLITS  64
#define LSPLIT  (L / SPLITS)   // 512

// Scheduling pins: batch VMEM reads ahead of WMMAs inside rolled loops so the
// wait inserter can use partial s_wait_loadcnt values (pipelined) while the
// rolled loop keeps register pressure bounded.
#if defined(__HIP_DEVICE_COMPILE__) && __has_builtin(__builtin_amdgcn_sched_group_barrier)
#define SCHED_LOADS_THEN_WMMA(nloads, nwmma)                  \
  do {                                                        \
    __builtin_amdgcn_sched_group_barrier(0x020, (nloads), 0); \
    __builtin_amdgcn_sched_group_barrier(0x008, (nwmma), 0);  \
  } while (0)
#else
#define SCHED_LOADS_THEN_WMMA(nloads, nwmma) do {} while (0)
#endif

// ---- WMMA fragment helpers (contiguous K order; A and B match) ------------------
// A (16x32 bf16): lane row = lane&15; half = lane>>4 selects K half:
//   elems 0..15 <- A[row][k0 + half*16 .. +15]
__device__ __forceinline__ v16bf load_a(const bf16* __restrict__ row, int half, int k0) {
  return *(const v16bf*)(row + k0 + half * 16);
}
// B (32x16 bf16): lane col = lane&15; lanes 0-15 hold K=0..15, lanes 16-31
// K=16..31 -> one contiguous 32B load from B^T row.
__device__ __forceinline__ v16bf load_b(const bf16* __restrict__ btrow, int half, int k0) {
  return *(const v16bf*)(btrow + k0 + half * 16);
}
__device__ __forceinline__ v8f wmma_bf16(v16bf a, v16bf b, v8f c) {
  return __builtin_amdgcn_wmma_f32_16x16x32_bf16(false, a, false, b, (short)0, c, false, false);
}

// ---- small utility kernels -------------------------------------------------
__global__ void k_f32_to_bf16(const float* __restrict__ in, bf16* __restrict__ out, int n) {
  int i = blockIdx.x * blockDim.x + threadIdx.x;
  if (i < n) out[i] = (bf16)in[i];
}

// GroupNorm stats: one block per group (8 ch x 32768). Deterministic tree.
__global__ void k_gn_stats(const float* __restrict__ x,
                           float* __restrict__ mean, float* __restrict__ rstd) {
  __shared__ float s1[256], s2[256];
  const int g = blockIdx.x;
  const float* p = x + (size_t)g * CPG * L;
  const int n = CPG * L;
  float a = 0.f, b = 0.f;
  for (int i = threadIdx.x; i < n; i += 256) { float v = p[i]; a += v; b += v * v; }
  s1[threadIdx.x] = a; s2[threadIdx.x] = b; __syncthreads();
  for (int o = 128; o > 0; o >>= 1) {
    if (threadIdx.x < o) { s1[threadIdx.x] += s1[threadIdx.x + o];
                           s2[threadIdx.x] += s2[threadIdx.x + o]; }
    __syncthreads();
  }
  if (threadIdx.x == 0) {
    float mu = s1[0] / n;
    float var = s2[0] / n - mu * mu;
    mean[g] = mu;
    rstd[g] = rsqrtf(var + 1e-5f);
  }
}

// Normalize + transpose: x (256 x L, l-contig) -> xnT (L x 256, c-contig) bf16.
__global__ void k_norm_t(const float* __restrict__ x,
                         const float* __restrict__ mean, const float* __restrict__ rstd,
                         const float* __restrict__ gw, const float* __restrict__ gb,
                         bf16* __restrict__ xnT) {
  __shared__ float tile[32][33];
  const int c0 = blockIdx.x * 32, l0 = blockIdx.y * 32;
  const int tx = threadIdx.x, ty = threadIdx.y;
#pragma unroll
  for (int j = 0; j < 4; j++) {
    int c = c0 + ty + j * 8;
    float v = x[(size_t)c * L + l0 + tx];
    int g = c >> 3;
    tile[ty + j * 8][tx] = (v - mean[g]) * rstd[g] * gw[c] + gb[c];
  }
  __syncthreads();
#pragma unroll
  for (int j = 0; j < 4; j++) {
    int l = l0 + ty + j * 8;
    xnT[(size_t)l * DIM + c0 + tx] = (bf16)tile[tx][ty + j * 8];
  }
}

// ---- QKV GEMM: (1536x256) @ (256xL). A frags in regs for whole K. ----------
// Rolled t-loop; sched-group barrier batches 16 B loads ahead of 8 WMMAs.
// launch_bounds(128,1) lifts the VGPR budget so nothing spills.
__global__ void __launch_bounds__(128, 1)
k_qkv(const bf16* __restrict__ W, const bf16* __restrict__ xnT,
      const float* __restrict__ bqkv,
      bf16* __restrict__ qT, float* __restrict__ kbuf,
      bf16* __restrict__ vbuf) {
  const int lane = threadIdx.x & 31, wave = threadIdx.x >> 5;
  const int half = lane >> 4, nl = lane & 15;
  const int mt = blockIdx.x >> 6;                       // 0..95 (uniform per block)
  const int ng = ((blockIdx.x & 63) << 2) + wave;       // 0..255
  const int m0 = mt * 16;
  const bf16* arow = W + (size_t)(m0 + nl) * DIM;
  v16bf a[8];
#pragma unroll
  for (int s = 0; s < 8; s++) a[s] = load_a(arow, half, s * 32);
  float bias[8];
#pragma unroll
  for (int r = 0; r < 8; r++) bias[r] = bqkv[m0 + half * 8 + r];
  const int region = m0 >> 9;              // 0:q 1:k 2:v (scalar)
#pragma unroll 1
  for (int t = 0; t < 8; t++) {
    const int n0 = ng * 128 + t * 16;
    const bf16* brow = xnT + (size_t)(n0 + nl) * DIM;
    v16bf b[8];
#pragma unroll
    for (int s = 0; s < 8; s++) b[s] = load_b(brow, half, s * 32);
    v8f c = {};
#pragma unroll
    for (int s = 0; s < 8; s++) c = wmma_bf16(a[s], b[s], c);
    SCHED_LOADS_THEN_WMMA(16, 8);
    if (region == 0) {
      const int col = m0 + half * 8;
      v8bf pk;
#pragma unroll
      for (int r = 0; r < 8; r++) pk[r] = (bf16)(c[r] + bias[r]);
      *(v8bf*)(qT + (size_t)(n0 + nl) * HID + col) = pk;
    } else if (region == 1) {
      const int f0 = m0 - 512 + half * 8;
#pragma unroll
      for (int r = 0; r < 8; r++)
        kbuf[(size_t)(f0 + r) * L + n0 + nl] = c[r] + bias[r];
    } else {
      const int f0 = m0 - 1024 + half * 8;
#pragma unroll
      for (int r = 0; r < 8; r++)
        vbuf[(size_t)(f0 + r) * L + n0 + nl] = (bf16)(c[r] + bias[r]);
    }
  }
}

// ---- softmax over L per k-row: online (max,sum) then deterministic reduce --
__global__ void k_softmax_stats(const float* __restrict__ kbuf,
                                float* __restrict__ rmax, float* __restrict__ rsum) {
  __shared__ float sm[256], ss[256];
  const int row = blockIdx.x;
  const float* p = kbuf + (size_t)row * L;
  float m = -INFINITY, s = 0.f;
  for (int i = threadIdx.x; i < L; i += 256) {
    float v = p[i];
    if (v > m) { s = s * __expf(m - v) + 1.f; m = v; }
    else       { s += __expf(v - m); }
  }
  sm[threadIdx.x] = m; ss[threadIdx.x] = s; __syncthreads();
  for (int o = 128; o > 0; o >>= 1) {
    if (threadIdx.x < o) {
      float m2 = sm[threadIdx.x + o], s2 = ss[threadIdx.x + o];
      float M = fmaxf(sm[threadIdx.x], m2);
      ss[threadIdx.x] = ss[threadIdx.x] * __expf(sm[threadIdx.x] - M) + s2 * __expf(m2 - M);
      sm[threadIdx.x] = M;
    }
    __syncthreads();
  }
  if (threadIdx.x == 0) { rmax[row] = sm[0]; rsum[row] = ss[0]; }
}

// E = exp(k - rowmax) in bf16 (unnormalized; 1/rowsum folded into context).
__global__ void k_expw(const float* __restrict__ kbuf, const float* __restrict__ rmax,
                       bf16* __restrict__ E) {
  size_t i = ((size_t)blockIdx.x * 256 + threadIdx.x) * 4;
  const int row = (int)(i >> 15);          // / L (L = 2^15)
  const float m = rmax[row];
  float4 v = *(const float4*)(kbuf + i);
  v4bf o;
  o[0] = (bf16)__expf(v.x - m); o[1] = (bf16)__expf(v.y - m);
  o[2] = (bf16)__expf(v.z - m); o[3] = (bf16)__expf(v.w - m);
  *(v4bf*)(E + i) = o;
}

// ---- context partials: per head, D[c,e] = sum_l E[c,l] * v[e,l], split-L ---
// 2 rolled K-groups of 8 steps; 32-load batch pinned ahead of the 8 WMMAs.
__global__ void __launch_bounds__(128, 1)
k_ctx(const bf16* __restrict__ E, const bf16* __restrict__ vbuf,
      float* __restrict__ part) {
  const int lane = threadIdx.x & 31, wave = threadIdx.x >> 5;
  const int half = lane >> 4, nl = lane & 15;
  const int job = blockIdx.x * 4 + wave;   // SPLITS * 8 heads * 16 tiles
  const int sp = job >> 7;
  const int rem = job & 127;
  const int h = rem >> 4, mtile = (rem >> 2) & 3, ntile = rem & 3;
  const bf16* arow = E    + (size_t)(h * 64 + mtile * 16 + nl) * L + sp * LSPLIT;
  const bf16* brow = vbuf + (size_t)(h * 64 + ntile * 16 + nl) * L + sp * LSPLIT;
  v8f c = {};
#pragma unroll 1
  for (int g = 0; g < 2; g++) {
    v16bf af[8], bf_[8];
#pragma unroll
    for (int s = 0; s < 8; s++) {
      af[s]  = load_a(arow, half, (g * 8 + s) * 32);
      bf_[s] = load_b(brow, half, (g * 8 + s) * 32);
    }
#pragma unroll
    for (int s = 0; s < 8; s++) c = wmma_bf16(af[s], bf_[s], c);
    SCHED_LOADS_THEN_WMMA(32, 8);
  }
  float* dst = part + (size_t)(sp * 8 + h) * 4096;
#pragma unroll
  for (int r = 0; r < 8; r++)
    dst[(mtile * 16 + half * 8 + r) * 64 + ntile * 16 + nl] = c[r];
}

// Reduce split-L partials (fixed order -> deterministic) and apply 1/rowsum.
__global__ void k_ctx_reduce(const float* __restrict__ part, const float* __restrict__ rsum,
                             bf16* __restrict__ ctx) {
  const int idx = blockIdx.x * 256 + threadIdx.x;   // 8*64*64
  const int h = idx >> 12;
  const int c = (idx >> 6) & 63;
  float s = 0.f;
  for (int sp = 0; sp < SPLITS; sp++) s += part[(size_t)(sp * 8 + h) * 4096 + (idx & 4095)];
  ctx[idx] = (bf16)(s / rsum[h * 64 + c]);
}

// ---- W2[d, h*64+c] = sum_e w_out[d, h*64+e] * ctx[h,c,e] -------------------
__global__ void __launch_bounds__(128, 1)
k_w2(const bf16* __restrict__ Wo, const bf16* __restrict__ ctx,
     bf16* __restrict__ W2) {
  const int lane = threadIdx.x & 31, wave = threadIdx.x >> 5;
  const int half = lane >> 4, nl = lane & 15;
  const int job = blockIdx.x * 4 + wave;   // 8 heads * 16 m-tiles * 4 n-tiles
  const int h = job >> 6;
  const int rem = job & 63;
  const int mtile = rem >> 2, ntile = rem & 3;
  const bf16* arow = Wo  + (size_t)(mtile * 16 + nl) * HID + h * 64;
  const bf16* brow = ctx + (size_t)(h * 64 + ntile * 16 + nl) * 64;
  v16bf a0 = load_a(arow, half, 0),  a1 = load_a(arow, half, 32);
  v16bf b0 = load_b(brow, half, 0),  b1 = load_b(brow, half, 32);
  v8f c = {};
  c = wmma_bf16(a0, b0, c);
  c = wmma_bf16(a1, b1, c);
#pragma unroll
  for (int r = 0; r < 8; r++)
    W2[(size_t)(mtile * 16 + half * 8 + r) * HID + h * 64 + ntile * 16 + nl] = (bf16)c[r];
}

// ---- final: out = W2 (256x512) @ q (512xL) + b_out, B-frags from qT --------
// A (16 frags) in regs; K in 2 rolled batches of 8 B-frags per n-tile.
__global__ void __launch_bounds__(128, 1)
k_final(const bf16* __restrict__ W2, const bf16* __restrict__ qT,
        const float* __restrict__ bout, float* __restrict__ out) {
  const int lane = threadIdx.x & 31, wave = threadIdx.x >> 5;
  const int half = lane >> 4, nl = lane & 15;
  const int mt = blockIdx.x >> 6;                       // 0..15 (uniform per block)
  const int ng = ((blockIdx.x & 63) << 2) + wave;       // 0..255
  const int m0 = mt * 16;
  const bf16* arow = W2 + (size_t)(m0 + nl) * HID;
  v16bf a[16];
#pragma unroll
  for (int s = 0; s < 16; s++) a[s] = load_a(arow, half, s * 32);
  float bias[8];
#pragma unroll
  for (int r = 0; r < 8; r++) bias[r] = bout[m0 + half * 8 + r];
#pragma unroll 1
  for (int t = 0; t < 8; t++) {
    const int n0 = ng * 128 + t * 16;
    const bf16* brow = qT + (size_t)(n0 + nl) * HID;
    v8f c = {};
#pragma unroll 1
    for (int g = 0; g < 2; g++) {
      v16bf b[8];
#pragma unroll
      for (int s = 0; s < 8; s++) b[s] = load_b(brow, half, (g * 8 + s) * 32);
#pragma unroll
      for (int s = 0; s < 8; s++) c = wmma_bf16(a[g * 8 + s], b[s], c);
      SCHED_LOADS_THEN_WMMA(16, 8);
    }
#pragma unroll
    for (int r = 0; r < 8; r++)
      out[(size_t)(m0 + half * 8 + r) * L + n0 + nl] = c[r] + bias[r];
  }
}

// ---------------------------------------------------------------------------
extern "C" void kernel_launch(void* const* d_in, const int* in_sizes, int n_in,
                              void* d_out, int out_size, void* d_ws, size_t ws_size,
                              hipStream_t stream) {
  const float* x    = (const float*)d_in[0];
  const float* gnw  = (const float*)d_in[1];
  const float* gnb  = (const float*)d_in[2];
  const float* wqkv = (const float*)d_in[3];
  const float* bqkv = (const float*)d_in[4];
  const float* wout = (const float*)d_in[5];
  const float* bout = (const float*)d_in[6];
  float* out = (float*)d_out;
  (void)in_sizes; (void)n_in; (void)out_size; (void)ws_size;

  char* ws = (char*)d_ws;
  size_t off = 0;
  auto alloc = [&](size_t bytes) {
    char* p = ws + off;
    off = (off + bytes + 255) & ~(size_t)255;
    return p;
  };
  bf16*  xnT  = (bf16*) alloc((size_t)L * DIM * 2);          // 16 MB
  bf16*  wqbf = (bf16*) alloc((size_t)3 * HID * DIM * 2);    // 768 KB
  bf16*  wobf = (bf16*) alloc((size_t)DIM * HID * 2);        // 256 KB
  bf16*  qT   = (bf16*) alloc((size_t)L * HID * 2);          // 32 MB
  float* kbuf = (float*)alloc((size_t)HID * L * 4);          // 64 MB
  bf16*  vbuf = (bf16*) alloc((size_t)HID * L * 2);          // 32 MB
  bf16*  Ebuf = (bf16*) alloc((size_t)HID * L * 2);          // 32 MB
  float* rmax = (float*)alloc(HID * 4);
  float* rsum = (float*)alloc(HID * 4);
  float* mean = (float*)alloc(GROUPS * 4);
  float* rstd = (float*)alloc(GROUPS * 4);
  float* part = (float*)alloc((size_t)SPLITS * HEADS * 64 * 64 * 4);  // 8 MB
  bf16*  ctx  = (bf16*) alloc((size_t)HEADS * 64 * 64 * 2);
  bf16*  W2   = (bf16*) alloc((size_t)DIM * HID * 2);

  k_f32_to_bf16<<<(3 * HID * DIM) / 256, 256, 0, stream>>>(wqkv, wqbf, 3 * HID * DIM);
  k_f32_to_bf16<<<(DIM * HID) / 256, 256, 0, stream>>>(wout, wobf, DIM * HID);
  k_gn_stats<<<GROUPS, 256, 0, stream>>>(x, mean, rstd);
  k_norm_t<<<dim3(DIM / 32, L / 32), dim3(32, 8), 0, stream>>>(x, mean, rstd, gnw, gnb, xnT);
  k_qkv<<<96 * 64, 128, 0, stream>>>(wqbf, xnT, bqkv, qT, kbuf, vbuf);
  k_softmax_stats<<<HID, 256, 0, stream>>>(kbuf, rmax, rsum);
  k_expw<<<(HID * (L / 4)) / 256, 256, 0, stream>>>(kbuf, rmax, Ebuf);
  k_ctx<<<(SPLITS * HEADS * 16) / 4, 128, 0, stream>>>(Ebuf, vbuf, part);
  k_ctx_reduce<<<(HEADS * 64 * 64) / 256, 256, 0, stream>>>(part, rsum, ctx);
  k_w2<<<(HEADS * 16 * 4) / 4, 128, 0, stream>>>(wobf, ctx, W2);
  k_final<<<16 * 64, 128, 0, stream>>>(W2, qT, bout, out);
}
